// DarkMachinesMixtureModel_46523085750821
// MI455X (gfx1250) — matrix operand order repaired and checked by reading, not compiled, not run on hardware
//
#include <hip/hip_runtime.h>
#include <hip/hip_bf16.h>
#include <math.h>

typedef __attribute__((ext_vector_type(16))) _Float16 v16h;
typedef __attribute__((ext_vector_type(8)))  _Float16 v8h;
typedef __attribute__((ext_vector_type(8)))  float    v8f;

#define FD     34
#define HIDP   288
#define KCAT   192
#define EMBD   128
#define OUTP   784
#define NBATCH 65536
#define ROWS   16
#define NTHR   288

// LDS strides (elements), chosen 16B-aligned per row and low-conflict
#define SA   296   // f16 activation buffer stride (592B rows, 148 DW: 148%64=20 -> conflict-free b128)
#define SC   136   // f16 ctx stride (272B rows)
#define SH   292   // f32 hidden stride
#define SO   792   // f16 spline-param stride

struct LayerW {
  const _Float16 *Wcat0, *W1[2], *W2[2], *Wc[2], *Wf;
  const float *b0, *b1[2], *b2[2], *bc[2], *bf;
  const int *perm;
};
struct KParams {
  const float *x; const int *dig; const int *c2i; const float *emb;
  float *out;
  LayerW L[2];
};

// ================= weight packing kernels (masks folded in, f16, padded) =================

__global__ void pack_cat0(_Float16* __restrict__ dst, float* __restrict__ bdst,
                          const float* __restrict__ W0, const float* __restrict__ Wc0,
                          const float* __restrict__ b0, const float* __restrict__ bc0) {
  int i = blockIdx.x * blockDim.x + threadIdx.x;
  if (i < HIDP * KCAT) {
    int n = i / KCAT, k = i - n * KCAT;
    float v = 0.f;
    if (n < 272) {
      if (k < FD) { if ((n % 33) >= k) v = W0[n * FD + k]; }       // M0: hid_deg>=in_deg
      else if (k >= 64) v = Wc0[n * EMBD + (k - 64)];
    }
    dst[i] = (_Float16)v;
  }
  if (i < HIDP) bdst[i] = (i < 272) ? (b0[i] + bc0[i]) : 0.f;
}

__global__ void pack_hh(_Float16* __restrict__ dst, const float* __restrict__ W) {
  int i = blockIdx.x * blockDim.x + threadIdx.x;
  if (i >= HIDP * HIDP) return;
  int n = i / HIDP, k = i - n * HIDP;
  float v = 0.f;
  if (n < 272 && k < 272 && (n % 33) >= (k % 33)) v = W[n * 272 + k];  // Mh
  dst[i] = (_Float16)v;
}

__global__ void pack_ctx(_Float16* __restrict__ dst, const float* __restrict__ W) {
  int i = blockIdx.x * blockDim.x + threadIdx.x;
  if (i >= HIDP * EMBD) return;
  int n = i / EMBD, k = i - n * EMBD;
  dst[i] = (_Float16)((n < 272) ? W[n * EMBD + k] : 0.f);
}

__global__ void pack_f(_Float16* __restrict__ dst, const float* __restrict__ W) {
  int i = blockIdx.x * blockDim.x + threadIdx.x;
  if (i >= OUTP * HIDP) return;
  int o = i / HIDP, k = i - o * HIDP;
  float v = 0.f;
  if (o < 782 && k < 272 && (o / 23) > (k % 33)) v = W[o * 272 + k];   // Mf: out_deg>hid_deg
  dst[i] = (_Float16)v;
}

__global__ void pack_bias(float* __restrict__ dst, const float* __restrict__ src,
                          int nsrc, int ndst) {
  int i = blockIdx.x * blockDim.x + threadIdx.x;
  if (i < ndst) dst[i] = (i < nsrc) ? src[i] : 0.f;
}

// ================= WMMA tile GEMM =================
// A (16xK f16) from LDS: lane holds row=lane%16, K-base (lane/16)*16, two 8-half chunks (+0, +8).
// B (Kx16 f16) from global [N][K]: lane holds col N=tile16+lane%16, 16 contiguous K at (lane/16)*16.
template<int K>
__device__ __forceinline__ v8f gemm_tile(const _Float16* aBase, const _Float16* bBase) {
  v8f c = {};
#pragma unroll
  for (int k = 0; k < K; k += 32) {
    v8h lo = *(const v8h*)(aBase + k);
    v8h hi = *(const v8h*)(aBase + k + 8);
    v16h a = __builtin_shufflevector(lo, hi, 0,1,2,3,4,5,6,7,8,9,10,11,12,13,14,15);
    v16h b = *(const v16h*)(bBase + k);
    c = __builtin_amdgcn_wmma_f32_16x16x32_f16(false, a, false, b, (short)0, c, false, false);
  }
  return c;
}

// ================= rational-quadratic spline =================
__device__ __forceinline__ float softplusf(float z) {
  return (z > 15.f) ? z : log1pf(expf(z));
}
__device__ __forceinline__ float pick9(const float* a, int i) {
  float r = a[0];
#pragma unroll
  for (int j = 1; j < 9; ++j) r = (i >= j) ? a[j] : r;
  return r;
}
__device__ __forceinline__ void rqs_one(const _Float16* tpRow, float x,
                                        float& yOut, float& ladOut) {
  const float s = 0.06063390625908324f;  // 1/sqrt(272)
  float tp[23];
#pragma unroll
  for (int j = 0; j < 23; ++j) tp[j] = (float)tpRow[j];

  float cw[9], ch[9];
  {
    float m = -1e30f;
#pragma unroll
    for (int i = 0; i < 8; ++i) m = fmaxf(m, tp[i] * s);
    float ex[8], sum = 0.f;
#pragma unroll
    for (int i = 0; i < 8; ++i) { ex[i] = expf(tp[i] * s - m); sum += ex[i]; }
    float inv = 1.f / sum, acc = 0.f;
    cw[0] = -5.f;
#pragma unroll
    for (int i = 0; i < 8; ++i) { acc += 0.001f + 0.992f * ex[i] * inv; cw[i + 1] = 10.f * acc - 5.f; }
    cw[8] = 5.f;
  }
  {
    float m = -1e30f;
#pragma unroll
    for (int i = 0; i < 8; ++i) m = fmaxf(m, tp[8 + i] * s);
    float ex[8], sum = 0.f;
#pragma unroll
    for (int i = 0; i < 8; ++i) { ex[i] = expf(tp[8 + i] * s - m); sum += ex[i]; }
    float inv = 1.f / sum, acc = 0.f;
    ch[0] = -5.f;
#pragma unroll
    for (int i = 0; i < 8; ++i) { acc += 0.001f + 0.992f * ex[i] * inv; ch[i + 1] = 10.f * acc - 5.f; }
    ch[8] = 5.f;
  }
  float dd[9];
  dd[0] = 1.f; dd[8] = 1.f;   // MIN_D + softplus(DERIV_PAD) == 1 exactly
#pragma unroll
  for (int i = 1; i <= 7; ++i) dd[i] = 0.001f + softplusf(tp[15 + i]);  // ud is NOT scaled

  bool inside = (x >= -5.f) && (x <= 5.f);
  float xc = fminf(fmaxf(x, -5.f), 5.f);
  int idx = -1;
#pragma unroll
  for (int j = 0; j < 9; ++j) idx += (xc >= cw[j]) ? 1 : 0;
  idx = idx < 0 ? 0 : (idx > 7 ? 7 : idx);

  float icw = pick9(cw, idx), cwn = pick9(cw, idx + 1);
  float ich = pick9(ch, idx), chn = pick9(ch, idx + 1);
  float d0  = pick9(dd, idx), d1  = pick9(dd, idx + 1);
  float iw = cwn - icw, ih = chn - ich;
  float delta = ih / iw;
  float th = (xc - icw) / iw;
  float th1m = th * (1.f - th);
  float num = ih * (delta * th * th + d0 * th1m);
  float den = delta + (d0 + d1 - 2.f * delta) * th1m;
  float yv = ich + num / den;
  float omt = 1.f - th;
  float dnum = delta * delta * (d1 * th * th + 2.f * delta * th1m + d0 * omt * omt);
  float lad = logf(dnum) - 2.f * logf(den);
  yOut = inside ? yv : x;
  ladOut = inside ? lad : 0.f;
}

// ================= fused MAF kernel: 16 rows / workgroup, 9 waves =================
__global__ __launch_bounds__(NTHR, 1) void maf_kernel(KParams P) {
  __shared__ __attribute__((aligned(16))) _Float16 aA[ROWS][SA];
  __shared__ __attribute__((aligned(16))) _Float16 ctxB[ROWS][SC];
  __shared__ __attribute__((aligned(16))) float    hB[ROWS][SH];
  __shared__ __attribute__((aligned(16))) _Float16 outB[ROWS][SO];
  __shared__ float xrow[ROWS][FD + 2];
  __shared__ float ladM[ROWS][FD];
  __shared__ int   permS[FD];
  __shared__ float ladB[ROWS];
  __shared__ int   idxS[ROWS];

  const int tid  = threadIdx.x;
  const int row0 = blockIdx.x * ROWS;
  const int lane = tid & 31;
  const int wave = tid >> 5;
  const int lrow = lane & 15;
  const int lhi  = lane >> 4;

  // ---- phase 0: discrete codes, ctx gather, x load ----
  if (tid < ROWS) {
    int r = tid, code = 0, f = 1;
#pragma unroll
    for (int i = 0; i < 8; ++i) { code += P.dig[(row0 + r) * 8 + i] * f; f *= 8; }
    idxS[r] = P.c2i[code];
    ladB[r] = 0.f;
  }
  __syncthreads();
  for (int e = tid; e < ROWS * EMBD; e += NTHR) {
    int r = e >> 7, c = e & 127;
    int idx = idxS[r]; if (idx < 0) idx = 0;
    ctxB[r][c] = (_Float16)P.emb[(size_t)idx * EMBD + c];
  }
  for (int e = tid; e < ROWS * FD; e += NTHR) {
    int r = e / FD, d = e - r * FD;
    xrow[r][d] = P.x[(size_t)(row0 + r) * FD + d];
  }
  __syncthreads();

  for (int l = 0; l < 2; ++l) {
    const LayerW lw = P.L[l];
    if (tid < FD) permS[tid] = lw.perm[tid];
    __syncthreads();

    // ---- permute x (register-staged), build A = [xp | 0 | ctx] (K=192) ----
    int e0 = tid, e1 = tid + NTHR;
    float v0 = 0.f, v1 = 0.f;
    if (e0 < ROWS * FD) v0 = xrow[e0 / FD][permS[e0 % FD]];
    if (e1 < ROWS * FD) v1 = xrow[e1 / FD][permS[e1 % FD]];
    __syncthreads();
    if (e0 < ROWS * FD) { int r = e0 / FD, d = e0 - r * FD; xrow[r][d] = v0; aA[r][d] = (_Float16)v0; }
    if (e1 < ROWS * FD) { int r = e1 / FD, d = e1 - r * FD; xrow[r][d] = v1; aA[r][d] = (_Float16)v1; }
    for (int e = tid; e < ROWS * 30; e += NTHR) aA[e / 30][FD + e % 30] = (_Float16)0.f;
    for (int e = tid; e < ROWS * EMBD; e += NTHR) { int r = e >> 7, c = e & 127; aA[r][64 + c] = ctxB[r][c]; }
    __syncthreads();

    // ---- GEMM1: h = [xp|ctx] @ Wcat0^T + (b0+bc0), 18 N-tiles over 9 waves ----
#pragma unroll
    for (int ti = 0; ti < 2; ++ti) {
      int n = (wave * 2 + ti) * 16 + lrow;
      v8f c = gemm_tile<KCAT>(&aA[lrow][lhi * 16], lw.Wcat0 + (size_t)n * KCAT + lhi * 16);
      float bn = lw.b0[n];
#pragma unroll
      for (int r = 0; r < 8; ++r) hB[r + 8 * lhi][n] = c[r] + bn;
    }
    __syncthreads();

    // ---- residual blocks ----
    for (int blk = 0; blk < 2; ++blk) {
      const _Float16* W1 = lw.W1[blk];
      const _Float16* W2 = lw.W2[blk];
      const _Float16* Wc = lw.Wc[blk];
      const float *b1 = lw.b1[blk], *b2 = lw.b2[blk], *bc = lw.bc[blk];

#pragma unroll
      for (int r = 0; r < ROWS; ++r) aA[r][tid] = (_Float16)fmaxf(hB[r][tid], 0.f);
      __syncthreads();

      v8f c1[2];
#pragma unroll
      for (int ti = 0; ti < 2; ++ti) {
        int n = (wave * 2 + ti) * 16 + lrow;
        c1[ti] = gemm_tile<HIDP>(&aA[lrow][lhi * 16], W1 + (size_t)n * HIDP + lhi * 16);
      }
      __syncthreads();                 // all reads of aA done before overwrite
#pragma unroll
      for (int ti = 0; ti < 2; ++ti) {
        int n = (wave * 2 + ti) * 16 + lrow;
        float bn = b1[n];
#pragma unroll
        for (int r = 0; r < 8; ++r) aA[r + 8 * lhi][n] = (_Float16)fmaxf(c1[ti][r] + bn, 0.f);
      }
      __syncthreads();
#pragma unroll
      for (int ti = 0; ti < 2; ++ti) {
        int n = (wave * 2 + ti) * 16 + lrow;
        v8f c2 = gemm_tile<HIDP>(&aA[lrow][lhi * 16], W2 + (size_t)n * HIDP + lhi * 16);
        v8f cc = gemm_tile<EMBD>(&ctxB[lrow][lhi * 16], Wc + (size_t)n * EMBD + lhi * 16);
        float b2n = b2[n], bcn = bc[n];
#pragma unroll
        for (int r = 0; r < 8; ++r) {
          float tcur = c2[r] + b2n;
          float sg = 1.f / (1.f + expf(-(cc[r] + bcn)));
          hB[r + 8 * lhi][n] += tcur * sg;
        }
      }
      __syncthreads();
    }

    // ---- final projection: out = relu(h) @ Wf^T + bf (49 N-tiles strided over waves) ----
#pragma unroll
    for (int r = 0; r < ROWS; ++r) aA[r][tid] = (_Float16)fmaxf(hB[r][tid], 0.f);
    __syncthreads();
    for (int nt = wave; nt < OUTP / 16; nt += 9) {
      int n = nt * 16 + lrow;
      v8f c = gemm_tile<HIDP>(&aA[lrow][lhi * 16], lw.Wf + (size_t)n * HIDP + lhi * 16);
      float bn = lw.bf[n];
#pragma unroll
      for (int r = 0; r < 8; ++r) outB[r + 8 * lhi][n] = (_Float16)(c[r] + bn);
    }
    __syncthreads();

    // ---- spline per (row, dim) ----
#pragma unroll
    for (int pass = 0; pass < 2; ++pass) {
      int ee = tid + pass * NTHR;
      if (ee < ROWS * FD) {
        int r = ee / FD, d = ee - r * FD;
        float y, lad;
        rqs_one(&outB[r][d * 23], xrow[r][d], y, lad);
        xrow[r][d] = y;
        ladM[r][d] = lad;
      }
    }
    __syncthreads();
    if (tid < ROWS) {                  // deterministic per-row reduction
      float s = ladB[tid];
#pragma unroll
      for (int d = 0; d < FD; ++d) s += ladM[tid][d];
      ladB[tid] = s;
    }
    __syncthreads();
  }

  // ---- outputs: y (B*34) | out_discrete (B) | lad (B) ----
  for (int e = tid; e < ROWS * FD; e += NTHR) {
    int r = e / FD, d = e - r * FD;
    P.out[(size_t)(row0 + r) * FD + d] = xrow[r][d];
  }
  if (tid < ROWS) {
    int r = tid, idx = idxS[r];
    P.out[(size_t)NBATCH * FD + row0 + r] = (float)idx;   // -1 when unknown
    P.out[(size_t)NBATCH * FD + NBATCH + row0 + r] = (idx < 0) ? -INFINITY : ladB[r];
  }
}

// ================= host =================
extern "C" void kernel_launch(void* const* d_in, const int* in_sizes, int n_in,
                              void* d_out, int out_size, void* d_ws, size_t ws_size,
                              hipStream_t stream) {
  if (n_in < 42) return;

  const float* x   = (const float*)d_in[0];
  const int*   dig = (const int*)d_in[1];
  const int*   c2i = (const int*)d_in[2];
  const float* emb = (const float*)d_in[3];

  const float *W0[2], *Wc0[2], *Wf[2], *b0[2], *bc0[2], *bf[2];
  const float *W1[2][2], *W2[2][2], *Wc[2][2], *b1[2][2], *b2[2][2], *bc[2][2];

  // Detect pytree flatten order of flow_params: dict-keys-sorted (jax.tree) vs insertion.
  bool sortedMode = (in_sizes[5] == 34816);
  int p = 4;
  for (int l = 0; l < 2; ++l) {
    if (sortedMode) {   // W0, Wc0, Wf, b0, bc0, bf, blocks{W1,W2,Wc,b1,b2,bc}x2
      W0[l]  = (const float*)d_in[p + 0];
      Wc0[l] = (const float*)d_in[p + 1];
      Wf[l]  = (const float*)d_in[p + 2];
      b0[l]  = (const float*)d_in[p + 3];
      bc0[l] = (const float*)d_in[p + 4];
      bf[l]  = (const float*)d_in[p + 5];
      int q = p + 6;
      for (int b = 0; b < 2; ++b) {
        W1[l][b] = (const float*)d_in[q + 0];
        W2[l][b] = (const float*)d_in[q + 1];
        Wc[l][b] = (const float*)d_in[q + 2];
        b1[l][b] = (const float*)d_in[q + 3];
        b2[l][b] = (const float*)d_in[q + 4];
        bc[l][b] = (const float*)d_in[q + 5];
        q += 6;
      }
    } else {            // insertion: W0,b0,Wc0,bc0, blocks{W1,b1,W2,b2,Wc,bc}x2, Wf,bf
      W0[l]  = (const float*)d_in[p + 0];
      b0[l]  = (const float*)d_in[p + 1];
      Wc0[l] = (const float*)d_in[p + 2];
      bc0[l] = (const float*)d_in[p + 3];
      int q = p + 4;
      for (int b = 0; b < 2; ++b) {
        W1[l][b] = (const float*)d_in[q + 0];
        b1[l][b] = (const float*)d_in[q + 1];
        W2[l][b] = (const float*)d_in[q + 2];
        b2[l][b] = (const float*)d_in[q + 3];
        Wc[l][b] = (const float*)d_in[q + 4];
        bc[l][b] = (const float*)d_in[q + 5];
        q += 6;
      }
      Wf[l] = (const float*)d_in[p + 16];
      bf[l] = (const float*)d_in[p + 17];
    }
    p += 18;
  }
  const int* perm0 = (const int*)d_in[40];
  const int* perm1 = (const int*)d_in[41];

  // workspace layout (256B aligned chunks)
  char* ws = (char*)d_ws;
  size_t off = 0;
  auto take = [&](size_t bytes) -> char* {
    char* pc = ws + off;
    off += (bytes + 255) & ~(size_t)255;
    return pc;
  };

  KParams KP;
  KP.x = x; KP.dig = dig; KP.c2i = c2i; KP.emb = emb; KP.out = (float*)d_out;

  for (int l = 0; l < 2; ++l) {
    _Float16* wcat = (_Float16*)take((size_t)HIDP * KCAT * 2);
    float*    b0c  = (float*)take((size_t)HIDP * 4);
    pack_cat0<<<(HIDP * KCAT + 255) / 256, 256, 0, stream>>>(wcat, b0c, W0[l], Wc0[l], b0[l], bc0[l]);
    KP.L[l].Wcat0 = wcat; KP.L[l].b0 = b0c;

    for (int b = 0; b < 2; ++b) {
      _Float16* w1 = (_Float16*)take((size_t)HIDP * HIDP * 2);
      _Float16* w2 = (_Float16*)take((size_t)HIDP * HIDP * 2);
      _Float16* wc = (_Float16*)take((size_t)HIDP * EMBD * 2);
      float* pb1 = (float*)take((size_t)HIDP * 4);
      float* pb2 = (float*)take((size_t)HIDP * 4);
      float* pbc = (float*)take((size_t)HIDP * 4);
      pack_hh<<<(HIDP * HIDP + 255) / 256, 256, 0, stream>>>(w1, W1[l][b]);
      pack_hh<<<(HIDP * HIDP + 255) / 256, 256, 0, stream>>>(w2, W2[l][b]);
      pack_ctx<<<(HIDP * EMBD + 255) / 256, 256, 0, stream>>>(wc, Wc[l][b]);
      pack_bias<<<2, 256, 0, stream>>>(pb1, b1[l][b], 272, HIDP);
      pack_bias<<<2, 256, 0, stream>>>(pb2, b2[l][b], 272, HIDP);
      pack_bias<<<2, 256, 0, stream>>>(pbc, bc[l][b], 272, HIDP);
      KP.L[l].W1[b] = w1; KP.L[l].W2[b] = w2; KP.L[l].Wc[b] = wc;
      KP.L[l].b1[b] = pb1; KP.L[l].b2[b] = pb2; KP.L[l].bc[b] = pbc;
    }

    _Float16* wf  = (_Float16*)take((size_t)OUTP * HIDP * 2);
    float*    pbf = (float*)take((size_t)OUTP * 4);
    pack_f<<<(OUTP * HIDP + 255) / 256, 256, 0, stream>>>(wf, Wf[l]);
    pack_bias<<<4, 256, 0, stream>>>(pbf, bf[l], 782, OUTP);
    KP.L[l].Wf = wf; KP.L[l].bf = pbf;
    KP.L[l].perm = (l == 0) ? perm0 : perm1;
  }

  maf_kernel<<<NBATCH / ROWS, NTHR, 0, stream>>>(KP);
}
